// ShiftReduceDP_76407468196365
// MI455X (gfx1250) — compile-verified, matrix operand order checked by reading, not run on hardware
//
#include <hip/hip_runtime.h>
#include <hip/hip_bf16.h>

// ---------------------------------------------------------------------------
// ShiftReduce inside-algorithm DP with neural pair scorers, CDNA5 (gfx1250).
// Heavy GEMM ([16192,512]@[512,8192], 16136 valid rows) runs on v_wmma_f32_16x16x32_bf16 with a
// fused online logsumexp epilogue (logits never materialized).  B tile is
// double-buffered in LDS with register-staged global loads; WMMA fragment
// loads are software-pipelined against the matrix ops.
// ---------------------------------------------------------------------------

#define L_   64
#define B_   8
#define H_   512
#define V_   8192
#define P_   2016          // pairs i<j
#define NP_  2017          // + synthetic pair (0,0)
#define NROWS_ 16136       // NP_*B_
#define MPAD_  16192       // padded to multiple of 64
#define NEGV  (-1e9f)
#define EPSV  4.5399929762484854e-5f   // exp(-10)

typedef __attribute__((ext_vector_type(16))) __bf16 v16bf;
typedef __attribute__((ext_vector_type(8)))  float  v8f;

// ---------------- K0: pair index tables -------------------------------------
__global__ void k0_pairs(int* ii, int* jj) {
  int p = blockIdx.x * blockDim.x + threadIdx.x;
  if (p >= NP_) return;
  if (p == P_) { ii[p] = 0; jj[p] = 0; return; }   // synthetic (0,0)
  int c = p, i = 0, run = L_ - 1;
  while (c >= run) { c -= run; ++i; --run; }
  ii[p] = i; jj[p] = i + 1 + c;
}

// ---------------- K1: W2w -> bf16 copy + fp32 transpose ---------------------
__global__ void k1_conv(const float* __restrict__ w, __bf16* __restrict__ o,
                        float* __restrict__ wt) {
  for (int idx = blockIdx.x * blockDim.x + threadIdx.x; idx < H_ * V_;
       idx += gridDim.x * blockDim.x) {
    const int k = idx >> 13;          // idx / V_
    const int n = idx & (V_ - 1);
    const float v = w[idx];
    o[idx] = (__bf16)v;
    wt[n * H_ + k] = v;               // W2wT[n][k] for coalesced gather
  }
}

// ---------------- K2: hidden-layer GEMM (fp32, tiny) ------------------------
// z: 0->At(W1t top) 1->Ab(W1t bot) 2->Awt(W1w top) 3->Awb(W1w bot)
__global__ void k2_hidden(const float* __restrict__ enc,
                          const float* __restrict__ W1t,
                          const float* __restrict__ W1w,
                          float* At, float* Ab, float* Awt, float* Awb) {
  __shared__ float sA[16][17], sB[16][17];
  const int tx = threadIdx.x, ty = threadIdx.y;
  const int row = blockIdx.y * 16 + ty;
  const int col = blockIdx.x * 16 + tx;
  const int z = blockIdx.z;
  const float* W = (z < 2) ? W1t : W1w;
  const int koff = (z & 1) ? H_ : 0;
  float* out = (z == 0) ? At : (z == 1) ? Ab : (z == 2) ? Awt : Awb;
  float acc = 0.f;
  for (int t = 0; t < H_ / 16; ++t) {
    sA[ty][tx] = enc[row * H_ + t * 16 + tx];
    sB[ty][tx] = W[(koff + t * 16 + ty) * H_ + col];
    __syncthreads();
#pragma unroll
    for (int k = 0; k < 16; ++k) acc += sA[ty][k] * sB[k][tx];
    __syncthreads();
  }
  out[row * H_ + col] = acc;
}

// ---------------- K3: transition scores (sigmoid -> sh/rr logs) -------------
__global__ void k3_trans(const int* __restrict__ ii, const int* __restrict__ jj,
                         const float* __restrict__ At, const float* __restrict__ Ab,
                         const float* __restrict__ b1t, const float* __restrict__ W2t,
                         const float* __restrict__ b2t,
                         float* __restrict__ sh, float* __restrict__ rr) {
  const int g = blockIdx.x * 8 + (threadIdx.x >> 5);   // wave id: (pair,batch)
  const int lane = threadIdx.x & 31;
  const int p = g >> 3, b = g & 7;                     // p < 2016 by grid size
  const int i = ii[p], j = jj[p];
  const float* ai = At + (i * B_ + b) * H_;
  const float* aj = Ab + (j * B_ + b) * H_;
  float part = 0.f;
  for (int k = lane; k < H_; k += 32) {
    float h = fmaxf(ai[k] + aj[k] + b1t[k], 0.f);
    part += h * W2t[k];
  }
#pragma unroll
  for (int off = 16; off >= 1; off >>= 1) part += __shfl_xor(part, off, 32);
  if (lane == 0) {
    const float sc = part + b2t[0];
    const float tp = 1.f / (1.f + __expf(-sc));
    sh[(i * L_ + j) * B_ + b] = log1pf(-tp + EPSV);
    rr[(i * L_ + j) * B_ + b] = __logf(tp + EPSV);
  }
}

// ---------------- K4: build word-hidden rows (bf16) -------------------------
__global__ void k4_hw(const int* __restrict__ ii, const int* __restrict__ jj,
                      const float* __restrict__ Awt, const float* __restrict__ Awb,
                      const float* __restrict__ b1w, __bf16* __restrict__ Hw) {
  for (int idx = blockIdx.x * blockDim.x + threadIdx.x; idx < MPAD_ * H_;
       idx += gridDim.x * blockDim.x) {
    const int r = idx >> 9, k = idx & (H_ - 1);
    float v = 0.f;
    if (r < NROWS_) {
      const int p = r >> 3, b = r & 7;
      const int i = ii[p], j = jj[p];
      v = fmaxf(Awt[(i * B_ + b) * H_ + k] + Awb[(j * B_ + b) * H_ + k] + b1w[k], 0.f);
    }
    Hw[idx] = (__bf16)v;
  }
}

// ---------------- K5: WMMA GEMM + fused online logsumexp --------------------
// 16-bit A fragment layout (ISA 7.12.2): lanes 0-15 row=lane, vgpr v holds
// K = (v&3)*2 + half*8 + (v>>2)*16 (half = lane>=16), pairs (K,K+1).
__device__ inline v16bf ld_frag(const __bf16* base, int stride) {
  const int lane = threadIdx.x & 31;
  const int row  = lane & 15;
  const int half = lane >> 4;
  v16bf f;
#pragma unroll
  for (int v = 0; v < 8; ++v) {
    const int kb = ((v & 3) << 1) + (half << 3) + ((v >> 2) << 4);
    f[2 * v]     = base[row * stride + kb];
    f[2 * v + 1] = base[row * stride + kb + 1];
  }
  return f;
}

#define MT 64
#define NT 32
#define NITER (V_ / NT)    // 256
__global__ void __launch_bounds__(256)
k5_gemm_lse(const __bf16* __restrict__ Hw, const __bf16* __restrict__ Wb,
            const float* __restrict__ b2w, float* __restrict__ lse) {
  __shared__ __align__(16) __bf16 sA[MT * H_];        // 64 KB (Hw rows)
  __shared__ __align__(16) __bf16 sB[2][NT * H_];     // 2x32 KB, [n][k] layout
  const int tid  = threadIdx.x;
  const int lane = tid & 31;
  const int wave = tid >> 5;
  const int wm = wave & 3;     // 4 M-subtiles
  const int wn = wave >> 2;    // 2 N-subtiles
  const int rowBase = blockIdx.x * MT;

  // A tile (contiguous chunk of Hw) -> LDS
  {
    const unsigned* src = (const unsigned*)(Hw + (size_t)rowBase * H_);
    unsigned* dst = (unsigned*)sA;
    for (int idx = tid; idx < MT * H_ / 2; idx += 256) dst[idx] = src[idx];
  }

  // Each thread stages rows k0,k0+1 of the 512x32 B slice (16B vector loads),
  // then scatters into LDS transposed as paired-k 32-bit stores.
  const int k0 = tid * 2;
  uint4 r0[4], r1[4];
  auto stage_load = [&](int n0) {
    const uint4* p0 = (const uint4*)(Wb + (size_t)k0 * V_ + n0);
    const uint4* p1 = (const uint4*)(Wb + (size_t)(k0 + 1) * V_ + n0);
#pragma unroll
    for (int c = 0; c < 4; ++c) { r0[c] = p0[c]; r1[c] = p1[c]; }
  };
  auto stage_store = [&](__bf16* dstB) {
    unsigned* d = (unsigned*)dstB;
#pragma unroll
    for (int c = 0; c < 4; ++c) {
      const unsigned u0[4] = {r0[c].x, r0[c].y, r0[c].z, r0[c].w};
      const unsigned u1[4] = {r1[c].x, r1[c].y, r1[c].z, r1[c].w};
#pragma unroll
      for (int q = 0; q < 4; ++q) {
        const int n = c * 8 + q * 2;
        d[(n * H_ + k0) >> 1]       = (u0[q] & 0xffffu) | (u1[q] << 16);
        d[((n + 1) * H_ + k0) >> 1] = (u0[q] >> 16) | (u1[q] & 0xffff0000u);
      }
    }
  };

  float m[8], s[8];
#pragma unroll
  for (int r = 0; r < 8; ++r) { m[r] = -1e30f; s[r] = 0.f; }
  const int myn = (wn << 4) + (lane & 15);   // my column (mod NT)

  stage_load(0);
  stage_store(sB[0]);
  __syncthreads();

  for (int it = 0; it < NITER; ++it) {
    const __bf16* bs = sB[it & 1];
    if (it + 1 < NITER) stage_load((it + 1) * NT);   // overlap with compute

    const __bf16* abase = sA + (wm << 4) * H_;
    const __bf16* bbase = bs + (wn << 4) * H_;
    v8f acc = {};
    v16bf ac = ld_frag(abase, H_);
    v16bf bc = ld_frag(bbase, H_);
#pragma unroll
    for (int kk = 0; kk < H_ / 32; ++kk) {
      v16bf an = {}, bn = {};
      if (kk + 1 < H_ / 32) {            // pipeline: fetch next frags pre-WMMA
        an = ld_frag(abase + (kk + 1) * 32, H_);
        bn = ld_frag(bbase + (kk + 1) * 32, H_);
      }
      acc = __builtin_amdgcn_wmma_f32_16x16x32_bf16(false, ac, false, bc,
                                                    (short)0, acc, false, false);
      ac = an; bc = bn;
    }
    const float bias = b2w[it * NT + myn];
#pragma unroll
    for (int r = 0; r < 8; ++r) {
      const float v = acc[r] + bias;
      const float mn = fmaxf(m[r], v);
      s[r] = s[r] * __expf(m[r] - mn) + __expf(v - mn);
      m[r] = mn;
    }
    if (it + 1 < NITER) stage_store(sB[(it + 1) & 1]);
    __syncthreads();
  }

  // reduce across the 16 lanes that share a row (halves are independent)
#pragma unroll
  for (int r = 0; r < 8; ++r) {
#pragma unroll
    for (int off = 8; off >= 1; off >>= 1) {
      const float om = __shfl_xor(m[r], off, 32);
      const float os = __shfl_xor(s[r], off, 32);
      const float mn = fmaxf(m[r], om);
      s[r] = s[r] * __expf(m[r] - mn) + os * __expf(om - mn);
      m[r] = mn;
    }
  }
  // combine the two wn-waves covering the same rows, via LDS
  float* red = (float*)sB[0];   // reuse: 256 floats
  if ((lane & 15) == 0) {
    const int half = lane >> 4;
    float* dst = red + (((wm * 2 + wn) * 2 + half) * 8) * 2;
#pragma unroll
    for (int r = 0; r < 8; ++r) { dst[2 * r] = m[r]; dst[2 * r + 1] = s[r]; }
  }
  __syncthreads();
  if (tid < 64) {
    const int lm = tid >> 4, half = (tid >> 3) & 1, r = tid & 7;
    const float m0 = red[(((lm * 2 + 0) * 2 + half) * 8 + r) * 2 + 0];
    const float s0 = red[(((lm * 2 + 0) * 2 + half) * 8 + r) * 2 + 1];
    const float m1 = red[(((lm * 2 + 1) * 2 + half) * 8 + r) * 2 + 0];
    const float s1 = red[(((lm * 2 + 1) * 2 + half) * 8 + r) * 2 + 1];
    const float mn = fmaxf(m0, m1);
    const float ss = s0 * __expf(m0 - mn) + s1 * __expf(m1 - mn);
    const int row = rowBase + lm * 16 + half * 8 + r;
    if (row < NROWS_) lse[row] = mn + __logf(ss);
  }
}

// ---------------- K6: gather target word log-prob (coalesced via W2wT) ------
__global__ void k6_gather(const int* __restrict__ ii, const int* __restrict__ jj,
                          const int* __restrict__ sent,
                          const float* __restrict__ Awt, const float* __restrict__ Awb,
                          const float* __restrict__ b1w, const float* __restrict__ W2wT,
                          const float* __restrict__ b2w, const float* __restrict__ lse,
                          float* __restrict__ gtab) {
  const int g = blockIdx.x * 8 + (threadIdx.x >> 5);
  if (g >= NROWS_) return;
  const int lane = threadIdx.x & 31;
  const int p = g >> 3, b = g & 7;
  const int i = ii[p], j = jj[p];
  if (j > 62) return;                       // only base cells need g
  const int w = sent[(j + 1) * B_ + b];
  const float* ai = Awt + (i * B_ + b) * H_;
  const float* aj = Awb + (j * B_ + b) * H_;
  const float* wc = W2wT + (size_t)w * H_;  // contiguous column of W2w
  float part = 0.f;
  for (int k = lane; k < H_; k += 32) {
    const float h = fmaxf(ai[k] + aj[k] + b1w[k], 0.f);
    part += h * wc[k];
  }
#pragma unroll
  for (int off = 16; off >= 1; off >>= 1) part += __shfl_xor(part, off, 32);
  if (lane == 0)
    gtab[(i * L_ + j) * B_ + b] = part + b2w[w] - lse[g];
}

// ---------------- K7: init table + base cells -------------------------------
__global__ void k7a_fill(float* __restrict__ table) {
  for (int i = blockIdx.x * blockDim.x + threadIdx.x; i < L_ * L_ * L_ * B_;
       i += gridDim.x * blockDim.x)
    table[i] = NEGV;
}
__global__ void k7b_base(const float* __restrict__ sh, const float* __restrict__ gtab,
                         float* __restrict__ table) {
  const int idx = blockIdx.x * blockDim.x + threadIdx.x;
  if (idx >= L_ * L_ * B_) return;
  const int b = idx & 7, j = (idx >> 3) & 63, i = idx >> 9;
  if (i == 0 && j == 0) {
    table[1 * B_ + b] = gtab[b];                       // table[0,0,1]
  } else if (i < j && j <= 62) {
    table[((i * L_ + j) * L_ + (j + 1)) * B_ + b] =
        sh[(i * L_ + j) * B_ + b] + gtab[(i * L_ + j) * B_ + b];
  }
}

// ---------------- K8: inside recursion, one launch per gap ------------------
__global__ void k8_gap(float* __restrict__ table, const float* __restrict__ rr,
                       int gap) {
  const int I = L_ - gap;
  const int total = L_ * I * B_;
  const int idx = blockIdx.x * blockDim.x + threadIdx.x;
  if (idx >= total) return;
  const int b = idx & 7;
  const int rest = idx >> 3;
  const int iv = rest % I;
  const int l = rest / I;
  const int j = iv + gap;
  float m = -1e30f, s = 0.f;
  for (int k = iv + 1; k < j; ++k) {
    const float t = table[((l * L_ + iv) * L_ + k) * B_ + b]
                  + table[((iv * L_ + k) * L_ + j) * B_ + b]
                  + rr[(k * L_ + j) * B_ + b];
    const float mn = fmaxf(m, t);
    s = s * __expf(m - mn) + __expf(t - mn);
    m = mn;
  }
  table[((l * L_ + iv) * L_ + j) * B_ + b] = m + __logf(s);
}

// ---------------- K9: emit result -------------------------------------------
__global__ void k9_out(const float* __restrict__ table, float* __restrict__ out) {
  const int b = threadIdx.x;
  if (b < B_) out[b] = table[63 * B_ + b];   // table[0,0,63]
}

// ---------------------------------------------------------------------------
extern "C" void kernel_launch(void* const* d_in, const int* in_sizes, int n_in,
                              void* d_out, int out_size, void* d_ws, size_t ws_size,
                              hipStream_t stream) {
  (void)in_sizes; (void)n_in; (void)out_size; (void)ws_size;
  const float* enc  = (const float*)d_in[0];
  const int*   sent = (const int*)  d_in[1];
  const float* W1t  = (const float*)d_in[2];
  const float* b1t  = (const float*)d_in[3];
  const float* W2t  = (const float*)d_in[4];
  const float* b2t  = (const float*)d_in[5];
  const float* W1w  = (const float*)d_in[6];
  const float* b1w  = (const float*)d_in[7];
  const float* W2w  = (const float*)d_in[8];
  const float* b2w  = (const float*)d_in[9];
  float* out = (float*)d_out;

  char* ws = (char*)d_ws;
  size_t off = 0;
  auto alloc = [&](size_t bytes) { char* p = ws + off; off += (bytes + 255) & ~size_t(255); return p; };
  float*  At    = (float*) alloc(L_ * B_ * H_ * 4);
  float*  Ab    = (float*) alloc(L_ * B_ * H_ * 4);
  float*  Awt   = (float*) alloc(L_ * B_ * H_ * 4);
  float*  Awb   = (float*) alloc(L_ * B_ * H_ * 4);
  float*  sh    = (float*) alloc(L_ * L_ * B_ * 4);
  float*  rr    = (float*) alloc(L_ * L_ * B_ * 4);
  float*  gtab  = (float*) alloc(L_ * L_ * B_ * 4);
  float*  lse   = (float*) alloc(MPAD_ * 4);
  int*    iiA   = (int*)   alloc(NP_ * 4);
  int*    jjA   = (int*)   alloc(NP_ * 4);
  float*  table = (float*) alloc((size_t)L_ * L_ * L_ * B_ * 4);
  __bf16* W2wb  = (__bf16*)alloc((size_t)H_ * V_ * 2);
  float*  W2wT  = (float*) alloc((size_t)H_ * V_ * 4);
  __bf16* Hw    = (__bf16*)alloc((size_t)MPAD_ * H_ * 2);

  k0_pairs<<<8, 256, 0, stream>>>(iiA, jjA);
  k1_conv<<<2048, 256, 0, stream>>>(W2w, W2wb, W2wT);
  k2_hidden<<<dim3(32, 32, 4), dim3(16, 16), 0, stream>>>(enc, W1t, W1w, At, Ab, Awt, Awb);
  k3_trans<<<P_, 256, 0, stream>>>(iiA, jjA, At, Ab, b1t, W2t, b2t, sh, rr);
  k4_hw<<<2048, 256, 0, stream>>>(iiA, jjA, Awt, Awb, b1w, Hw);
  k5_gemm_lse<<<MPAD_ / MT, 256, 0, stream>>>(Hw, W2wb, b2w, lse);
  k6_gather<<<NP_, 256, 0, stream>>>(iiA, jjA, sent, Awt, Awb, b1w, W2wT, b2w, lse, gtab);
  k7a_fill<<<1024, 256, 0, stream>>>(table);
  k7b_base<<<(L_ * L_ * B_ + 255) / 256, 256, 0, stream>>>(sh, gtab, table);
  for (int gap = 2; gap <= L_ - 1; ++gap) {
    const int total = L_ * (L_ - gap) * B_;
    k8_gap<<<(total + 255) / 256, 256, 0, stream>>>(table, rr, gap);
  }
  k9_out<<<1, 32, 0, stream>>>(table, out);
}
